// LocalDWConvAttention_89962384982326
// MI455X (gfx1250) — matrix operand order repaired and checked by reading, not compiled, not run on hardware
//
#include <hip/hip_runtime.h>
#include <hip/hip_bf16.h>

// Problem constants (from reference setup_inputs)
constexpr int Bn = 8;
constexpr int C0 = 256;
constexpr int HH = 64;
constexpr int WW = 256;

typedef __attribute__((ext_vector_type(16))) __bf16          v16bf;
typedef __attribute__((ext_vector_type(16))) unsigned short  v16u;
typedef __attribute__((ext_vector_type(8)))  float           v8f;

__device__ __forceinline__ unsigned short f2bf(float f) {
  union { float f; unsigned u; } x; x.f = f;
  unsigned u = x.u;
  u += 0x7FFFu + ((u >> 16) & 1u);   // round-to-nearest-even
  return (unsigned short)(u >> 16);
}

// ---------------------------------------------------------------------------
// Pre-pack pointwise weights (fp32 O-major) into bf16 WMMA B-fragments.
// B-matrix 32x16 bf16 layout (ISA 7.12.2): lane l holds column n=l%16,
// element e holds K = 8*(l/16) + e + (e&8)  (within a 32-K chunk).
// Fragment f = (co_tile * KC + k_chunk), 512 elements, lane-major (l*16+e).
// Layer bases in fragments: L1:0 (8x8=64), L2:64 (4x4=16), L3:80 (2x2=4), L4:84 (1).
// ---------------------------------------------------------------------------
__global__ __launch_bounds__(256) void pack_pw_weights(
    const float* __restrict__ pw1, const float* __restrict__ pw2,
    const float* __restrict__ pw3, const float* __restrict__ pw4,
    unsigned short* __restrict__ packed) {
  int idx = blockIdx.x * 256 + threadIdx.x;
  const int TOTAL = 85 * 512;
  if (idx >= TOTAL) return;
  int f = idx >> 9;
  int r = idx & 511;
  int l = r >> 4, e = r & 15;
  const float* w; int Cin, Co, t, q;
  if (f < 64)      { int fl = f;      Cin = 256; Co = 128; w = pw1; t = fl >> 3; q = fl & 7; }
  else if (f < 80) { int fl = f - 64; Cin = 128; Co = 64;  w = pw2; t = fl >> 2; q = fl & 3; }
  else if (f < 84) { int fl = f - 80; Cin = 64;  Co = 32;  w = pw3; t = fl >> 1; q = fl & 1; }
  else             {                  Cin = 32;  Co = 9;   w = pw4; t = 0;       q = 0;      }
  int n = 16 * t + (l & 15);
  int k = 32 * q + 8 * (l >> 4) + e + (e & 8);
  float v = (n < Co) ? w[n * Cin + k] : 0.0f;
  packed[idx] = f2bf(v);
}

// ---------------------------------------------------------------------------
// Fused depthwise 3x3 (+bias) -> pointwise GEMM (bf16 WMMA, f32 accum)
// (+bias, optional ReLU).  One block = 8 rows x 16 cols tile; wave w = row w,
// M = 16 columns.  Output written NHWC (stride CO_STORE).
// B fragments for the current K-chunk are staged in LDS via the gfx1250
// async-to-LDS DMA path (ASYNCcnt), overlapping with x-tile staging and the
// depthwise VALU work.
// ---------------------------------------------------------------------------
template <int CIN, int CO, int CO_STORE, bool RELU, bool IN_NCHW>
__global__ __launch_bounds__(256) void dsconv_layer(
    const float* __restrict__ in, const float* __restrict__ dw_w,
    const float* __restrict__ dw_b, const float* __restrict__ pw_b,
    const unsigned short* __restrict__ bpack, float* __restrict__ out) {
  constexpr int NT = CO / 16;   // output channel tiles
  constexpr int KC = CIN / 32;  // 32-channel K chunks

  __shared__ float xt[32 * 181];                          // 10x18 halo tile x 32ch
  __shared__ alignas(32) unsigned short As[8 * 32 * 16];  // A fragments [wave][lane][16]
  __shared__ alignas(32) unsigned short Bs[NT * 512];     // B fragments for current kq

  const int tid = threadIdx.x;
  const int bid = blockIdx.x;
  const int tx  = bid & 15;        // W/16 = 16 column tiles
  const int ty  = (bid >> 4) & 7;  // H/8  = 8 row tiles
  const int b   = bid >> 7;
  const int row0 = ty * 8, col0 = tx * 16;

  const int wv = tid >> 5;   // wave id == tile row
  const int ln = tid & 31;

  // depthwise scatter constants: this thread owns channel c_ch of each chunk
  const int c_ch = tid & 31;
  const int pg   = tid >> 5;                 // row handled in dw phase
  const int gA   = (c_ch >> 3) & 1;          // lane-half for this K
  const int rem  = c_ch - 8 * gA;            // {0..7} U {16..23}
  const int jA   = (rem < 8) ? (rem >> 1) : (4 + ((rem - 16) >> 1));
  const int eA   = 2 * jA + (rem & 1);       // element index within v16

  v8f acc[NT];
  for (int t = 0; t < NT; ++t)
    for (int i = 0; i < 8; ++i) acc[t][i] = 0.0f;

  for (int kq = 0; kq < KC; ++kq) {
    __syncthreads();  // protect LDS reuse across iterations

    // ---- kick off async DMA of this K-chunk's B fragments into LDS ----
    // NT fragments of 1KB each; 16B (b128) per lane per iteration.
#pragma unroll
    for (int i = tid; i < NT * 64; i += 256) {
      int t = i >> 6;          // fragment (co tile)
      int u = i & 63;          // 16B unit within fragment
      unsigned           dst = (unsigned)(size_t)(&Bs[t * 512 + u * 8]);
      unsigned long long src = (unsigned long long)(bpack + (size_t)(t * KC + kq) * 512 + u * 8);
      asm volatile("global_load_async_to_lds_b128 %0, %1, off"
                   :: "v"(dst), "v"(src) : "memory");
    }

    // ---- stage 10x18 x 32ch input tile (zero pad at borders) ----
    if (IN_NCHW) {
      for (int i = tid; i < 5760; i += 256) {
        int c  = i / 180;
        int p  = i - c * 180;
        int ly = p / 18, lx = p - ly * 18;
        int y  = row0 - 1 + ly, xx = col0 - 1 + lx;
        float v = 0.0f;
        if ((unsigned)y < (unsigned)HH && (unsigned)xx < (unsigned)WW)
          v = in[(((size_t)b * CIN + (kq * 32 + c)) * HH + y) * WW + xx];
        xt[c * 181 + ly * 18 + lx] = v;
      }
    } else {  // NHWC: channels fastest -> coalesced
      for (int i = tid; i < 5760; i += 256) {
        int c  = i & 31;
        int p  = i >> 5;
        int ly = p / 18, lx = p - ly * 18;
        int y  = row0 - 1 + ly, xx = col0 - 1 + lx;
        float v = 0.0f;
        if ((unsigned)y < (unsigned)HH && (unsigned)xx < (unsigned)WW)
          v = in[(((size_t)b * HH + y) * WW + xx) * CIN + (kq * 32 + c)];
        xt[c * 181 + ly * 18 + lx] = v;
      }
    }
    __syncthreads();

    // ---- depthwise 3x3 + bias, scatter bf16 into A-fragment layout ----
    {
      const int ch = kq * 32 + c_ch;
      float kw[9];
#pragma unroll
      for (int i = 0; i < 9; ++i) kw[i] = dw_w[ch * 9 + i];
      const float bia = dw_b[ch];
      const float* xc = &xt[c_ch * 181];
#pragma unroll
      for (int ii = 0; ii < 16; ++ii) {   // ii = column within tile (== M index)
        float a = bia;
#pragma unroll
        for (int dy = 0; dy < 3; ++dy) {
          const float* xr = xc + (pg + dy) * 18 + ii;
          a = fmaf(xr[0], kw[dy * 3 + 0], a);
          a = fmaf(xr[1], kw[dy * 3 + 1], a);
          a = fmaf(xr[2], kw[dy * 3 + 2], a);
        }
        int l = ii + 16 * gA;
        As[(pg * 32 + l) * 16 + eA] = f2bf(a);
      }
    }

    // ensure this thread's async B copies have landed in LDS, then sync block
    asm volatile("s_wait_asynccnt 0x0" ::: "memory");
    __syncthreads();

    // ---- WMMA: A (dw outputs) x B (staged pw weights), both from LDS ----
    v16u au = *(const v16u*)&As[(wv * 32 + ln) * 16];
    v16bf av = __builtin_bit_cast(v16bf, au);
#pragma unroll
    for (int t = 0; t < NT; ++t) {
      v16u bu = *(const v16u*)&Bs[t * 512 + ln * 16];
      v16bf bv = __builtin_bit_cast(v16bf, bu);
      acc[t] = __builtin_amdgcn_wmma_f32_16x16x32_bf16(
          false, av, false, bv, (short)0, acc[t], false, false);
    }
  }

  // ---- epilogue: +pw bias, ReLU, NHWC store ----
  // C/D layout: lane holds column n = ln%16; VGPR r holds row m = r + 8*(ln/16)
  const int gD = ln >> 4, n0 = ln & 15;
  const int row = row0 + wv;
#pragma unroll
  for (int t = 0; t < NT; ++t) {
    int o = t * 16 + n0;
    if (o < CO_STORE) {
      float bias = pw_b[o];
#pragma unroll
      for (int r = 0; r < 8; ++r) {
        int col = col0 + r + 8 * gD;
        float v = acc[t][r] + bias;
        if (RELU) v = fmaxf(v, 0.0f);
        out[(((size_t)b * HH + row) * WW + col) * CO_STORE + o] = v;
      }
    }
  }
}

// ---------------------------------------------------------------------------
// Softmax over 9 logits + dynamic 3x3 conv over all 256 channels.
// One block = 8 rows x 32 cols; attn kept in registers (one position/thread).
// x halo tile staged in LDS per 4-channel chunk.
// ---------------------------------------------------------------------------
__global__ __launch_bounds__(256) void softmax_dynconv(
    const float* __restrict__ x, const float* __restrict__ logits,
    float* __restrict__ out, float* __restrict__ char_attn) {
  constexpr int CCH = 4;
  __shared__ float xs[CCH][10 * 35];

  const int bid = blockIdx.x;
  const int tx  = bid & 7;        // W/32 = 8
  const int ty  = (bid >> 3) & 7; // H/8  = 8
  const int b   = bid >> 6;
  const int row0 = ty * 8, col0 = tx * 32;

  const int tid = threadIdx.x;
  const int py = tid >> 5, px = tid & 31;
  const int y = row0 + py, xcol = col0 + px;

  // softmax over 9 taps (NHWC logits -> contiguous)
  const float* lg = logits + ((size_t)(b * HH + y) * WW + xcol) * 9;
  float a[9];
  float mx = lg[0];
#pragma unroll
  for (int i = 0; i < 9; ++i) { a[i] = lg[i]; mx = fmaxf(mx, a[i]); }
  float s = 0.0f;
#pragma unroll
  for (int i = 0; i < 9; ++i) { a[i] = __expf(a[i] - mx); s += a[i]; }
  const float inv = 1.0f / s;
  float* ca = char_attn + ((size_t)(b * HH + y) * WW + xcol) * 9;
#pragma unroll
  for (int i = 0; i < 9; ++i) { a[i] *= inv; ca[i] = a[i]; }

  for (int c0 = 0; c0 < C0; c0 += CCH) {
    __syncthreads();
    for (int i = tid; i < CCH * 340; i += 256) {
      int cc = i / 340;
      int p  = i - cc * 340;
      int ly = p / 34, lx = p - ly * 34;
      int yy = row0 - 1 + ly, xx = col0 - 1 + lx;
      float v = 0.0f;
      if ((unsigned)yy < (unsigned)HH && (unsigned)xx < (unsigned)WW)
        v = x[(((size_t)b * C0 + (c0 + cc)) * HH + yy) * WW + xx];
      xs[cc][ly * 35 + lx] = v;
    }
    __syncthreads();
#pragma unroll
    for (int cc = 0; cc < CCH; ++cc) {
      const float* xr = &xs[cc][0];
      float accv = 0.0f;
#pragma unroll
      for (int ki = 0; ki < 3; ++ki)
#pragma unroll
        for (int kj = 0; kj < 3; ++kj)
          accv = fmaf(xr[(py + ki) * 35 + (px + kj)], a[ki * 3 + kj], accv);
      out[(((size_t)b * C0 + (c0 + cc)) * HH + y) * WW + xcol] = accv;
    }
  }
}

// ---------------------------------------------------------------------------
extern "C" void kernel_launch(void* const* d_in, const int* in_sizes, int n_in,
                              void* d_out, int out_size, void* d_ws, size_t ws_size,
                              hipStream_t stream) {
  (void)in_sizes; (void)n_in; (void)out_size; (void)ws_size;

  const float* x     = (const float*)d_in[0];
  const float* dw1_w = (const float*)d_in[1];
  const float* dw1_b = (const float*)d_in[2];
  const float* pw1_w = (const float*)d_in[3];
  const float* pw1_b = (const float*)d_in[4];
  const float* dw2_w = (const float*)d_in[5];
  const float* dw2_b = (const float*)d_in[6];
  const float* pw2_w = (const float*)d_in[7];
  const float* pw2_b = (const float*)d_in[8];
  const float* dw3_w = (const float*)d_in[9];
  const float* dw3_b = (const float*)d_in[10];
  const float* pw3_w = (const float*)d_in[11];
  const float* pw3_b = (const float*)d_in[12];
  const float* dw4_w = (const float*)d_in[13];
  const float* dw4_b = (const float*)d_in[14];
  const float* pw4_w = (const float*)d_in[15];
  const float* pw4_b = (const float*)d_in[16];

  // Workspace layout (fp32 intermediates NHWC, then packed bf16 weights)
  float* ws  = (float*)d_ws;
  const size_t NP = (size_t)Bn * HH * WW;   // 131072 positions
  float* a1  = ws;                          // NP * 128
  float* a2  = a1 + NP * 128;               // NP * 64
  float* a3  = a2 + NP * 64;                // NP * 32
  float* lgt = a3 + NP * 32;                // NP * 9
  unsigned short* packed = (unsigned short*)(lgt + NP * 9);  // 85*512 bf16

  pack_pw_weights<<<(85 * 512 + 255) / 256, 256, 0, stream>>>(
      pw1_w, pw2_w, pw3_w, pw4_w, packed);

  const int nblk = Bn * (HH / 8) * (WW / 16);  // 1024
  dsconv_layer<256, 128, 128, true,  true ><<<nblk, 256, 0, stream>>>(
      x,  dw1_w, dw1_b, pw1_b, packed,            a1);
  dsconv_layer<128, 64,  64,  true,  false><<<nblk, 256, 0, stream>>>(
      a1, dw2_w, dw2_b, pw2_b, packed + 64 * 512, a2);
  dsconv_layer<64,  32,  32,  true,  false><<<nblk, 256, 0, stream>>>(
      a2, dw3_w, dw3_b, pw3_b, packed + 80 * 512, a3);
  dsconv_layer<32,  16,  9,   false, false><<<nblk, 256, 0, stream>>>(
      a3, dw4_w, dw4_b, pw4_b, packed + 84 * 512, lgt);

  float* outp = (float*)d_out;
  softmax_dynconv<<<Bn * (HH / 8) * (WW / 32), 256, 0, stream>>>(
      x, lgt, outp, outp + (size_t)Bn * C0 * HH * WW);
}
